// MambaBlock_77610059039066
// MI455X (gfx1250) — matrix operand order repaired and checked by reading, not compiled, not run on hardware
//
#include <hip/hip_runtime.h>
#include <hip/hip_bf16.h>
#include <math.h>

// ---------------------------------------------------------------------------
// MambaBlock for MI455X (gfx1250): bf16 WMMA GEMMs (v_wmma_f32_16x16x32_bf16),
// fp32 accumulation, bf16 staged intermediates (memory-bound workload:
// ~105 GFLOP vs ~0.7 GB traffic @ 23.3 TB/s).
// Pipeline:
//   0) convert x -> bf16; transpose+convert W_in, W_ssm, W_out -> bf16 [N,K]
//   1) GEMM1: h2 = x @ W_in + b_in, split -> xh_pre (bf16), silu(gate) (bf16)
//   2) depthwise causal conv K=4 + silu -> xh (bf16)
//   3) GEMM2: x_ssm = xh @ W_ssm + b_ssm (fp32, N=32)
//   4) scan: per-batch wave32 recurrence h = A h + B.*xs ; s_t = C.h  (general A)
//   5) elementwise: A2 = xh * (s_t + D_p) * silu(gate)  (bf16)
//   6) GEMM3: out = A2 @ W_out + b_out (fp32 -> d_out)
//
// Fragment layouts per CDNA5 ISA:
//   A (16-bit 16x32): lane L = row L&15; lanes 0-15 hold K {0..7, 16..23},
//                     lanes 16-31 hold K {8..15, 24..31}  (interleaved)
//   B (16-bit 32x16): lane L = col L&15; lanes 0-15 hold K {0..7, 16..23},
//                     lanes 16-31 hold K {8..15, 24..31}  (interleaved)
//   C/D (f32 16x16):  VGPR r, lane L -> (M = r + 8*(L>>4), N = L&15)
// ---------------------------------------------------------------------------

typedef __bf16 bf16_t;
typedef bf16_t v16bf __attribute__((ext_vector_type(16)));
typedef bf16_t v8bf  __attribute__((ext_vector_type(8)));
typedef float  v8f   __attribute__((ext_vector_type(8)));
typedef unsigned short v8us __attribute__((ext_vector_type(8)));

#define B_SZ 8
#define T_LEN 4096
#define D_IN 512
#define D_H 1024
#define D_S 32
#define KW 4
#define M_ROWS (B_SZ * T_LEN)   // 32768

__device__ __forceinline__ unsigned short f2bf(float f) {
    unsigned int u = __float_as_uint(f);
    unsigned int r = u + 0x7FFFu + ((u >> 16) & 1u);   // round-to-nearest-even
    return (unsigned short)(r >> 16);
}
__device__ __forceinline__ float bf2f(unsigned short h) {
    return __uint_as_float(((unsigned int)h) << 16);
}
__device__ __forceinline__ float silu(float x) {
    return x / (1.0f + __expf(-x));
}

union FragU {
    v16bf v;
    v8us  u[2];
};

// ---------------------------------------------------------------------------
// Prep kernels
// ---------------------------------------------------------------------------
__global__ void cvt_f32_bf16_kernel(const float* __restrict__ in,
                                    unsigned short* __restrict__ out, int n) {
    int i = blockIdx.x * blockDim.x + threadIdx.x;
    if (i < n) out[i] = f2bf(in[i]);
}

// in: [K,N] row-major fp32  ->  out: [N,K] row-major bf16
__global__ void transpose_bf16_kernel(const float* __restrict__ in,
                                      unsigned short* __restrict__ out,
                                      int K, int N) {
    int i = blockIdx.x * blockDim.x + threadIdx.x;
    if (i >= N * K) return;
    int n = i / K;
    int k = i - n * K;
    out[i] = f2bf(in[(size_t)k * N + n]);
}

// ---------------------------------------------------------------------------
// Main GEMM: C[M,N] = A[M,K](bf16) @ Bt[N,K](bf16)^T + bias
// Block: 256 thr = 8 waves (2 waveM x 4 waveN); block tile 128M x 128N.
// Wave tile: 64M x 32N = 4x2 WMMA 16x16 tiles, K step 32.
// EPI 0: split at n=1024: out0 <- bf16(v), out1 <- bf16(silu(v))  (GEMM1)
// EPI 1: outf[m*N+n] <- fp32(v)                                    (GEMM3)
// ---------------------------------------------------------------------------
template <int EPI>
__global__ __launch_bounds__(256) void gemm_bf16_kernel(
    const unsigned short* __restrict__ A,
    const unsigned short* __restrict__ Bt,
    const float* __restrict__ bias,
    int K, int N,
    unsigned short* __restrict__ out0,
    unsigned short* __restrict__ out1,
    float* __restrict__ outf) {

    const int lane = threadIdx.x & 31;
    const int wave = threadIdx.x >> 5;
    const int wm = wave >> 2;        // 0..1
    const int wn = wave & 3;         // 0..3
    const int r = lane & 15;
    const int half = lane >> 4;

    const int rowW = blockIdx.x * 128 + wm * 64;
    const int colW = blockIdx.y * 128 + wn * 32;

    v8f acc[4][2];
#pragma unroll
    for (int i = 0; i < 4; ++i)
#pragma unroll
        for (int j = 0; j < 2; ++j)
#pragma unroll
            for (int e = 0; e < 8; ++e) acc[i][j][e] = 0.0f;

    // A: interleaved per-lane K pattern (half*8, chunks at +0 / +16)
    const unsigned short* aBase = A + (size_t)(rowW + r) * K + half * 8;
    // B: interleaved per-lane K pattern (half*8, chunks at +0 / +16)
    const unsigned short* bBase = Bt + (size_t)(colW + r) * K + half * 8;

    for (int k0 = 0; k0 < K; k0 += 32) {
        FragU af[4], bf[2];
#pragma unroll
        for (int i = 0; i < 4; ++i) {
            const unsigned short* p = aBase + (size_t)i * 16 * K + k0;
            af[i].u[0] = *(const v8us*)(p);
            af[i].u[1] = *(const v8us*)(p + 16);
        }
#pragma unroll
        for (int j = 0; j < 2; ++j) {
            const unsigned short* p = bBase + (size_t)j * 16 * K + k0;
            bf[j].u[0] = *(const v8us*)(p);
            bf[j].u[1] = *(const v8us*)(p + 16);
        }
#pragma unroll
        for (int i = 0; i < 4; ++i)
#pragma unroll
            for (int j = 0; j < 2; ++j)
                acc[i][j] = __builtin_amdgcn_wmma_f32_16x16x32_bf16(
                    false, af[i].v, false, bf[j].v, (short)0, acc[i][j],
                    false, false);
    }

#pragma unroll
    for (int j = 0; j < 2; ++j) {
        const int n = colW + j * 16 + r;
        const float bv = bias[n];
#pragma unroll
        for (int i = 0; i < 4; ++i) {
#pragma unroll
            for (int e = 0; e < 8; ++e) {
                const int m = rowW + i * 16 + half * 8 + e;
                float v = acc[i][j][e] + bv;
                if (EPI == 0) {
                    if (n < D_H)
                        out0[(size_t)m * D_H + n] = f2bf(v);
                    else
                        out1[(size_t)m * D_H + (n - D_H)] = f2bf(silu(v));
                } else {
                    outf[(size_t)m * N + n] = v;
                }
            }
        }
    }
}

// ---------------------------------------------------------------------------
// Depthwise causal conv (K=4) + silu.  xh_pre bf16 -> xh bf16.
// ---------------------------------------------------------------------------
__global__ void conv_silu_kernel(const unsigned short* __restrict__ xh_pre,
                                 const float* __restrict__ conv_w,
                                 const float* __restrict__ conv_b,
                                 unsigned short* __restrict__ xh) {
    const int idx = blockIdx.x * blockDim.x + threadIdx.x;   // m*D_H + c
    const int m = idx >> 10;
    const int c = idx & (D_H - 1);
    const int t = m & (T_LEN - 1);
    float acc = conv_b[c];
#pragma unroll
    for (int k = 0; k < KW; ++k) {
        const int dt = t - (KW - 1) + k;
        if (dt >= 0)
            acc += conv_w[c * KW + k] *
                   bf2f(xh_pre[(size_t)(m - (KW - 1) + k) * D_H + c]);
    }
    xh[idx] = f2bf(silu(acc));
}

// ---------------------------------------------------------------------------
// GEMM2: x_ssm[M,32] = xh[M,1024] @ W_ssm + b_ssm  (N=32 -> 2 WMMA n-tiles).
// Block: 8 waves stacked in M; wave tile 16M x 32N; K=1024.
// ---------------------------------------------------------------------------
__global__ __launch_bounds__(256) void gemm_ssm_kernel(
    const unsigned short* __restrict__ A,     // xh [M, 1024]
    const unsigned short* __restrict__ Bt,    // W_ssm^T [32, 1024]
    const float* __restrict__ bias,           // b_ssm [32]
    float* __restrict__ out) {                // x_ssm [M, 32]
    const int lane = threadIdx.x & 31;
    const int wave = threadIdx.x >> 5;
    const int r = lane & 15;
    const int half = lane >> 4;
    const int rowW = blockIdx.x * 128 + wave * 16;
    const int K = D_H;

    v8f acc[2];
#pragma unroll
    for (int j = 0; j < 2; ++j)
#pragma unroll
        for (int e = 0; e < 8; ++e) acc[j][e] = 0.0f;

    const unsigned short* aBase = A + (size_t)(rowW + r) * K + half * 8;
    const unsigned short* bBase = Bt + (size_t)r * K + half * 8;

    for (int k0 = 0; k0 < K; k0 += 32) {
        FragU af, bf[2];
        af.u[0] = *(const v8us*)(aBase + k0);
        af.u[1] = *(const v8us*)(aBase + k0 + 16);
#pragma unroll
        for (int j = 0; j < 2; ++j) {
            const unsigned short* p = bBase + (size_t)j * 16 * K + k0;
            bf[j].u[0] = *(const v8us*)(p);
            bf[j].u[1] = *(const v8us*)(p + 16);
        }
#pragma unroll
        for (int j = 0; j < 2; ++j)
            acc[j] = __builtin_amdgcn_wmma_f32_16x16x32_bf16(
                false, af.v, false, bf[j].v, (short)0, acc[j], false, false);
    }

#pragma unroll
    for (int j = 0; j < 2; ++j) {
        const int n = j * 16 + r;
        const float bv = bias[n];
#pragma unroll
        for (int e = 0; e < 8; ++e) {
            const int m = rowW + half * 8 + e;
            out[(size_t)m * D_S + n] = acc[j][e] + bv;
        }
    }
}

// ---------------------------------------------------------------------------
// Sequential SSM scan (general A). One wave32 per batch; lane s owns h[s] and
// row s of A. Per step: h_new[s] = sum_j A[s][j]*h[j] + xs[s]*B_p[s];
// then s_t = sum_s C_p[s]*h[s] via xor-reduction. Stores one scalar per row.
// ---------------------------------------------------------------------------
__global__ __launch_bounds__(32) void ssm_scan_kernel(
    const float* __restrict__ x_ssm,   // [M, 32]
    const float* __restrict__ Amat,    // [32, 32]
    const float* __restrict__ B_p,     // [32]
    const float* __restrict__ C_p,     // [32]
    float* __restrict__ s_out) {       // [M]
    const int b = blockIdx.x;
    const int s = threadIdx.x;

    float arow[D_S];
#pragma unroll
    for (int j = 0; j < D_S; ++j) arow[j] = Amat[s * D_S + j];
    const float bp = B_p[s];
    const float cp = C_p[s];

    float h = 0.0f;
    const size_t base = (size_t)b * T_LEN;
    for (int t = 0; t < T_LEN; ++t) {
        const float xs = x_ssm[(base + t) * D_S + s];
        float hn = xs * bp;
#pragma unroll
        for (int j = 0; j < D_S; ++j) hn += arow[j] * __shfl(h, j, 32);
        h = hn;
        float u = cp * h;
#pragma unroll
        for (int off = 16; off >= 1; off >>= 1) u += __shfl_xor(u, off, 32);
        if (s == 0) s_out[base + t] = u;
    }
}

// ---------------------------------------------------------------------------
// A2[m,k] = xh[m,k] * (s[m] + D_p[k]) * silu_gate[m,k]   (bf16 out)
// ---------------------------------------------------------------------------
__global__ void build_a2_kernel(const unsigned short* __restrict__ xh,
                                const unsigned short* __restrict__ gate_silu,
                                const float* __restrict__ s_arr,
                                const float* __restrict__ D_p,
                                unsigned short* __restrict__ A2) {
    const int idx = blockIdx.x * blockDim.x + threadIdx.x;
    const int m = idx >> 10;
    const int k = idx & (D_H - 1);
    const float v = bf2f(xh[idx]) * (s_arr[m] + D_p[k]) * bf2f(gate_silu[idx]);
    A2[idx] = f2bf(v);
}

// ---------------------------------------------------------------------------
// Host launcher
// ---------------------------------------------------------------------------
extern "C" void kernel_launch(void* const* d_in, const int* in_sizes, int n_in,
                              void* d_out, int out_size, void* d_ws, size_t ws_size,
                              hipStream_t stream) {
    (void)in_sizes; (void)n_in; (void)out_size; (void)ws_size;

    const float* x      = (const float*)d_in[0];   // [8,4096,512]
    const float* W_in   = (const float*)d_in[1];   // [512,2048]
    const float* b_in   = (const float*)d_in[2];   // [2048]
    const float* conv_w = (const float*)d_in[3];   // [1024,1,4]
    const float* conv_b = (const float*)d_in[4];   // [1024]
    const float* W_ssm  = (const float*)d_in[5];   // [1024,32]
    const float* b_ssm  = (const float*)d_in[6];   // [32]
    const float* Amat   = (const float*)d_in[7];   // [32,32]
    const float* B_p    = (const float*)d_in[8];   // [32]
    const float* C_p    = (const float*)d_in[9];   // [32]
    const float* D_p    = (const float*)d_in[10];  // [1024]
    const float* W_out  = (const float*)d_in[11];  // [1024,512]
    const float* b_out  = (const float*)d_in[12];  // [512]
    float* out = (float*)d_out;                    // [8,4096,512]

    char* w = (char*)d_ws;
    auto alloc = [&](size_t bytes) -> char* {
        char* p = w;
        w += (bytes + 255) & ~(size_t)255;
        return p;
    };

    unsigned short* xB      = (unsigned short*)alloc((size_t)M_ROWS * D_IN * 2);
    unsigned short* WinT    = (unsigned short*)alloc((size_t)2 * D_H * D_IN * 2);
    unsigned short* WssmT   = (unsigned short*)alloc((size_t)D_S * D_H * 2);
    unsigned short* WoutT   = (unsigned short*)alloc((size_t)D_IN * D_H * 2);
    unsigned short* xh_pre  = (unsigned short*)alloc((size_t)M_ROWS * D_H * 2);
    unsigned short* gateS   = (unsigned short*)alloc((size_t)M_ROWS * D_H * 2);
    unsigned short* xh      = (unsigned short*)alloc((size_t)M_ROWS * D_H * 2);
    float*          xssm    = (float*)alloc((size_t)M_ROWS * D_S * 4);
    float*          sArr    = (float*)alloc((size_t)M_ROWS * 4);
    unsigned short* A2      = xh_pre;   // reuse: xh_pre dead after conv

    // 0) conversions / weight transposes
    {
        int n = M_ROWS * D_IN;
        cvt_f32_bf16_kernel<<<(n + 255) / 256, 256, 0, stream>>>(x, xB, n);
    }
    transpose_bf16_kernel<<<(2 * D_H * D_IN + 255) / 256, 256, 0, stream>>>(
        W_in, WinT, D_IN, 2 * D_H);
    transpose_bf16_kernel<<<(D_S * D_H + 255) / 256, 256, 0, stream>>>(
        W_ssm, WssmT, D_H, D_S);
    transpose_bf16_kernel<<<(D_IN * D_H + 255) / 256, 256, 0, stream>>>(
        W_out, WoutT, D_H, D_IN);

    // 1) GEMM1: h2 = x @ W_in + b_in, split -> xh_pre, silu(gate)
    gemm_bf16_kernel<0><<<dim3(M_ROWS / 128, (2 * D_H) / 128), 256, 0, stream>>>(
        xB, WinT, b_in, D_IN, 2 * D_H, xh_pre, gateS, nullptr);

    // 2) depthwise causal conv + silu
    conv_silu_kernel<<<(M_ROWS * D_H) / 256, 256, 0, stream>>>(
        xh_pre, conv_w, conv_b, xh);

    // 3) GEMM2: x_ssm = xh @ W_ssm + b_ssm
    gemm_ssm_kernel<<<M_ROWS / 128, 256, 0, stream>>>(xh, WssmT, b_ssm, xssm);

    // 4) sequential scan (general A), one wave per batch
    ssm_scan_kernel<<<B_SZ, 32, 0, stream>>>(xssm, Amat, B_p, C_p, sArr);

    // 5) A2 = xh * (s + D_p) * silu(gate)
    build_a2_kernel<<<(M_ROWS * D_H) / 256, 256, 0, stream>>>(
        xh, gateS, sArr, D_p, A2);

    // 6) GEMM3: out = A2 @ W_out + b_out
    gemm_bf16_kernel<1><<<dim3(M_ROWS / 128, D_IN / 128), 256, 0, stream>>>(
        A2, WoutT, b_out, D_H, D_IN, nullptr, nullptr, out);
}
